// ShiftedWindowAttention_8126078124479
// MI455X (gfx1250) — compile-verified, matrix operand order checked by reading, not run on hardware
//
#include <hip/hip_runtime.h>
#include <hip/hip_bf16.h>
#include <stdint.h>

// ---------------------------------------------------------------------------
// Fused shifted-window attention for MI455X (gfx1250, wave32, WMMA bf16).
// ---------------------------------------------------------------------------

typedef __attribute__((ext_vector_type(16))) __bf16        v16bf;
typedef __attribute__((ext_vector_type(8)))  float         v8f;
typedef __attribute__((ext_vector_type(4)))  unsigned int  u32x4;

union FragBF {
    v16bf v;
    u32x4 q[2];
};

__device__ __forceinline__ unsigned short f2bf(float f) {
    unsigned int u = __float_as_uint(f);
    u += 0x7FFFu + ((u >> 16) & 1u);   // round-to-nearest-even
    return (unsigned short)(u >> 16);
}

#define NTOK   36          // tokens per 6x6 window
#define MPAD   48          // padded to 3 M-tiles of 16
#define ROWP   264         // padded row stride (halfs) for C=256 rows
#define PSTR   64          // row stride (halfs) for P (48x64) and V^T (32x64)

// LDS byte offsets (dynamic shared)
#define OFF_X  0
#define SZ_X   (MPAD * ROWP * 2)         // 25344 : X window / later O tile
#define OFF_Q  (OFF_X + SZ_X)            // 25344 : Q rows (48 x 256)
#define SZ_Q   (MPAD * ROWP * 2)
#define OFF_K  (OFF_Q + SZ_Q)            // 50688 : K rows (48 x 256)
#define SZ_K   (MPAD * ROWP * 2)
#define OFF_P  OFF_Q                     // P aliases dead Q+K region
#define SZ_P   (8 * MPAD * PSTR * 2)     // 49152 : softmax probs, 8 heads
#define OFF_V  (OFF_K + SZ_K)            // 76032 : V^T per head (8 x 32 x 64)
#define SZ_V   (8 * 32 * PSTR * 2)       // 32768
#define LDS_BYTES (OFF_V + SZ_V)         // 108800

#define WMMA_BF16(A, B, Cv) \
    __builtin_amdgcn_wmma_f32_16x16x32_bf16(false, (A), false, (B), (short)0, (Cv), false, false)

// ---------------------------------------------------------------------------
// Prep: convert qkv_w (768x256) and proj_w (256x256) f32 -> bf16 into d_ws.
// Both are (N_out, K_in) row-major == B^T, exactly the WMMA B-frag layout.
// ---------------------------------------------------------------------------
__global__ void prep_weights(const float* __restrict__ qkv_w,
                             const float* __restrict__ proj_w,
                             unsigned short* __restrict__ wbf) {
    const int NQ = 768 * 256;
    const int NP = 256 * 256;
    int i = blockIdx.x * blockDim.x + threadIdx.x;
    if (i < NQ)            wbf[i] = f2bf(qkv_w[i]);
    else if (i < NQ + NP)  wbf[i] = f2bf(proj_w[i - NQ]);
}

// ---------------------------------------------------------------------------
// Fused kernel: one workgroup (8 wave32s) per shifted window.
// ---------------------------------------------------------------------------
__global__ __launch_bounds__(256)
void swin_fused(const float* __restrict__ x,
                const unsigned short* __restrict__ qwbf,   // 768x256 bf16
                const unsigned short* __restrict__ pwbf,   // 256x256 bf16
                const float* __restrict__ qkv_b,
                const float* __restrict__ proj_b,
                float* __restrict__ out) {
    extern __shared__ char smem[];
    unsigned short* sX = (unsigned short*)(smem + OFF_X);
    unsigned short* sQ = (unsigned short*)(smem + OFF_Q);
    unsigned short* sK = (unsigned short*)(smem + OFF_K);
    unsigned short* sP = (unsigned short*)(smem + OFF_P);
    unsigned short* sV = (unsigned short*)(smem + OFF_V);
    __shared__ int s_rid[MPAD];

    const int tid = threadIdx.x;
    const int wid = blockIdx.x;          // 0..8191
    const int b   = wid >> 8;
    const int w2  = wid & 255;
    const int wh6 = (w2 >> 4) * 6;
    const int ww6 = (w2 & 15) * 6;

    // Shift-window mask region ids (9 regions of the shifted image).
    if (tid < MPAD) {
        int r = -2;
        if (tid < NTOK) {
            int hr = wh6 + tid / 6;
            int wc = ww6 + tid % 6;
            int ih = (hr < 90) ? 0 : ((hr < 93) ? 1 : 2);
            int iw = (wc < 90) ? 0 : ((wc < 93) ? 1 : 2);
            r = ih * 3 + iw;
        }
        s_rid[tid] = r;
    }

    // Zero V^T (covers K-padding tokens 48..63 for the attn@V GEMM).
    {
        u32x4 z = {0u, 0u, 0u, 0u};
        u32x4* pv = (u32x4*)sV;
        for (int i = tid; i < SZ_V / 16; i += 256) pv[i] = z;
    }

    // Gather shifted window -> bf16 in LDS (rows 36..47 zero).
    {
        const int col = tid;  // 0..255
        for (int row = 0; row < MPAD; ++row) {
            unsigned short v = 0;
            if (row < NTOK) {
                int hh = wh6 + row / 6 + 3; if (hh >= 96) hh -= 96;
                int ww = ww6 + row % 6 + 3; if (ww >= 96) ww -= 96;
                v = f2bf(x[(((b * 96 + hh) * 96) + ww) * 256 + col]);
            }
            sX[row * ROWP + col] = v;
        }
    }
    __syncthreads();   // B1

    const int wv   = tid >> 5;          // wave id 0..7
    const int lane = tid & 31;
    const int lrow = lane & 15;         // M/N index within a 16-tile
    const int kb0  = (lane < 16) ? 0 : 8;  // K-chunk base per lane half

    // ---------------- Stage 2: QKV = X @ qkv_w^T + b -----------------------
    for (int j = 0; j < 6; ++j) {
        const int nt   = wv + j * 8;        // N-tile 0..47
        const int nout = nt * 16 + lrow;    // 0..767
        const unsigned short* wrow = qwbf + nout * 256;
        v8f acc[3];
        #pragma unroll
        for (int mt = 0; mt < 3; ++mt) acc[mt] = (v8f){0,0,0,0,0,0,0,0};
        #pragma unroll
        for (int ks = 0; ks < 8; ++ks) {
            const int kb = ks * 32 + kb0;
            FragBF B, A[3];
            B.q[0] = *(const u32x4*)(wrow + kb);
            B.q[1] = *(const u32x4*)(wrow + kb + 16);
            #pragma unroll
            for (int mt = 0; mt < 3; ++mt) {
                const unsigned short* ar = sX + (mt * 16 + lrow) * ROWP + kb;
                A[mt].q[0] = *(const u32x4*)(ar);
                A[mt].q[1] = *(const u32x4*)(ar + 16);
            }
            #pragma unroll
            for (int mt = 0; mt < 3; ++mt)
                acc[mt] = WMMA_BF16(A[mt].v, B.v, acc[mt]);
        }
        const float bias = qkv_b[nout];
        #pragma unroll
        for (int mt = 0; mt < 3; ++mt) {
            #pragma unroll
            for (int r = 0; r < 8; ++r) {
                const int m = mt * 16 + kb0 + r;
                const unsigned short bv = f2bf(acc[mt][r] + bias);
                if (nout < 256) {
                    sQ[m * ROWP + nout] = bv;
                } else if (nout < 512) {
                    sK[m * ROWP + (nout - 256)] = bv;
                } else {                       // V stored transposed per head
                    const int c = nout - 512;  // head = c>>5, d = c&31
                    sV[c * PSTR + m] = bv;     // (head*32 + d) * 64 + token
                }
            }
        }
    }
    __syncthreads();   // B2

    // ---------------- Stage 3: S = Q K^T, mask, softmax --------------------
    const int h = wv;  // one head per wave
    v8f S[3][3];
    {
        FragBF Aq[3], Bk[3];
        const int cb = h * 32 + kb0;
        #pragma unroll
        for (int t = 0; t < 3; ++t) {
            const unsigned short* qr = sQ + (t * 16 + lrow) * ROWP + cb;
            const unsigned short* kr = sK + (t * 16 + lrow) * ROWP + cb;
            Aq[t].q[0] = *(const u32x4*)(qr);
            Aq[t].q[1] = *(const u32x4*)(qr + 16);
            Bk[t].q[0] = *(const u32x4*)(kr);
            Bk[t].q[1] = *(const u32x4*)(kr + 16);
        }
        #pragma unroll
        for (int mt = 0; mt < 3; ++mt)
            #pragma unroll
            for (int nl = 0; nl < 3; ++nl) {
                v8f z = {0,0,0,0,0,0,0,0};
                S[mt][nl] = WMMA_BF16(Aq[mt].v, Bk[nl].v, z);
            }
    }
    __syncthreads();   // B3: Q/K reads done, P may overwrite that region

    {
        const float scale = 0.17677669529663687f;  // 1/sqrt(32)
        int ridn[3];
        #pragma unroll
        for (int nl = 0; nl < 3; ++nl) ridn[nl] = s_rid[nl * 16 + lrow];
        #pragma unroll
        for (int mt = 0; mt < 3; ++mt) {
            #pragma unroll
            for (int r = 0; r < 8; ++r) {
                const int m = mt * 16 + kb0 + r;
                const int ridm = s_rid[m];
                float v[3];
                #pragma unroll
                for (int nl = 0; nl < 3; ++nl) {
                    const int n = nl * 16 + lrow;
                    float s = S[mt][nl][r] * scale;
                    v[nl] = (n >= NTOK || ridn[nl] != ridm) ? -__builtin_inff() : s;
                }
                float mx = fmaxf(v[0], fmaxf(v[1], v[2]));
                #pragma unroll
                for (int d = 1; d < 16; d <<= 1) mx = fmaxf(mx, __shfl_xor(mx, d, 16));
                float e0 = __expf(v[0] - mx);
                float e1 = __expf(v[1] - mx);
                float e2 = __expf(v[2] - mx);
                float sum = e0 + e1 + e2;
                #pragma unroll
                for (int d = 1; d < 16; d <<= 1) sum += __shfl_xor(sum, d, 16);
                const float inv = 1.0f / (sum + 1e-20f);
                unsigned short* pr = sP + (h * MPAD + m) * PSTR;
                pr[0 * 16 + lrow] = f2bf(e0 * inv);
                pr[1 * 16 + lrow] = f2bf(e1 * inv);
                pr[2 * 16 + lrow] = f2bf(e2 * inv);
            }
        }
        // zero P K-padding columns 48..63
        u32x4 z4 = {0u, 0u, 0u, 0u};
        for (int rr = lane; rr < MPAD; rr += 32) {
            u32x4* pp = (u32x4*)(sP + (h * MPAD + rr) * PSTR + 48);
            pp[0] = z4; pp[1] = z4;
        }
    }
    __syncthreads();   // B4

    // ---------------- Stage 4: O = P @ V (per head) ------------------------
    {
        v8f O[3][2];
        #pragma unroll
        for (int mt = 0; mt < 3; ++mt)
            #pragma unroll
            for (int dt = 0; dt < 2; ++dt) O[mt][dt] = (v8f){0,0,0,0,0,0,0,0};
        #pragma unroll
        for (int ks = 0; ks < 2; ++ks) {
            const int kb = ks * 32 + kb0;
            FragBF Ap[3], Bv[2];
            #pragma unroll
            for (int mt = 0; mt < 3; ++mt) {
                const unsigned short* pr = sP + (h * MPAD + mt * 16 + lrow) * PSTR + kb;
                Ap[mt].q[0] = *(const u32x4*)(pr);
                Ap[mt].q[1] = *(const u32x4*)(pr + 16);
            }
            #pragma unroll
            for (int dt = 0; dt < 2; ++dt) {
                const unsigned short* vr = sV + (h * 32 + dt * 16 + lrow) * PSTR + kb;
                Bv[dt].q[0] = *(const u32x4*)(vr);
                Bv[dt].q[1] = *(const u32x4*)(vr + 16);
            }
            #pragma unroll
            for (int mt = 0; mt < 3; ++mt)
                #pragma unroll
                for (int dt = 0; dt < 2; ++dt)
                    O[mt][dt] = WMMA_BF16(Ap[mt].v, Bv[dt].v, O[mt][dt]);
        }
        // store O tile (bf16) into sX region (X is dead)
        #pragma unroll
        for (int mt = 0; mt < 3; ++mt)
            #pragma unroll
            for (int dt = 0; dt < 2; ++dt)
                #pragma unroll
                for (int r = 0; r < 8; ++r) {
                    const int m = mt * 16 + kb0 + r;
                    sX[m * ROWP + h * 32 + dt * 16 + lrow] = f2bf(O[mt][dt][r]);
                }
    }
    __syncthreads();   // B5

    // ---------------- Stage 5: out = O @ proj_w^T + b, inverse shift -------
    for (int j = 0; j < 2; ++j) {
        const int nt   = wv + j * 8;        // N-tile 0..15
        const int nout = nt * 16 + lrow;    // 0..255
        const unsigned short* wrow = pwbf + nout * 256;
        v8f acc[3];
        #pragma unroll
        for (int mt = 0; mt < 3; ++mt) acc[mt] = (v8f){0,0,0,0,0,0,0,0};
        #pragma unroll
        for (int ks = 0; ks < 8; ++ks) {
            const int kb = ks * 32 + kb0;
            FragBF B, A[3];
            B.q[0] = *(const u32x4*)(wrow + kb);
            B.q[1] = *(const u32x4*)(wrow + kb + 16);
            #pragma unroll
            for (int mt = 0; mt < 3; ++mt) {
                const unsigned short* ar = sX + (mt * 16 + lrow) * ROWP + kb;
                A[mt].q[0] = *(const u32x4*)(ar);
                A[mt].q[1] = *(const u32x4*)(ar + 16);
            }
            #pragma unroll
            for (int mt = 0; mt < 3; ++mt)
                acc[mt] = WMMA_BF16(A[mt].v, B.v, acc[mt]);
        }
        const float bias = proj_b[nout];
        #pragma unroll
        for (int mt = 0; mt < 3; ++mt) {
            #pragma unroll
            for (int r = 0; r < 8; ++r) {
                const int m = mt * 16 + kb0 + r;
                if (m < NTOK) {
                    int hh = wh6 + m / 6 + 3; if (hh >= 96) hh -= 96;
                    int wp = ww6 + m % 6 + 3; if (wp >= 96) wp -= 96;
                    out[(((b * 96 + hh) * 96) + wp) * 256 + nout] = acc[mt][r] + bias;
                }
            }
        }
    }
}

// ---------------------------------------------------------------------------
extern "C" void kernel_launch(void* const* d_in, const int* in_sizes, int n_in,
                              void* d_out, int out_size, void* d_ws, size_t ws_size,
                              hipStream_t stream) {
    (void)in_sizes; (void)n_in; (void)out_size; (void)ws_size;
    const float* x      = (const float*)d_in[0];
    const float* qkv_w  = (const float*)d_in[1];
    const float* qkv_b  = (const float*)d_in[2];
    const float* proj_w = (const float*)d_in[3];
    const float* proj_b = (const float*)d_in[4];
    float* out          = (float*)d_out;
    unsigned short* wbf = (unsigned short*)d_ws;   // 768*256 + 256*256 bf16

    const int NWCONV = 768 * 256 + 256 * 256;      // 262144 elements
    prep_weights<<<NWCONV / 256, 256, 0, stream>>>(qkv_w, proj_w, wbf);
    swin_fused<<<8192, 256, LDS_BYTES, stream>>>(
        x, wbf, wbf + 768 * 256, qkv_b, proj_b, out);
}
